// T10_transformer_70849780514963
// MI455X (gfx1250) — matrix operand order repaired and checked by reading, not compiled, not run on hardware
//
#include <hip/hip_runtime.h>
#include <hip/hip_bf16.h>

#define B_TOTAL 65536
#define T_STEPS 34

typedef __attribute__((ext_vector_type(16))) _Float16 v16h;
typedef __attribute__((ext_vector_type(8)))  _Float16 v8h;
typedef __attribute__((ext_vector_type(8)))  float    v8f;
typedef __attribute__((ext_vector_type(4)))  float    v4f;

static __device__ __forceinline__ v8f wmma_f16(v16h a, v16h b, v8f c) {
  // D = A(16x32 f16) * B(32x16 f16) + C(16x16 f32)
  return __builtin_amdgcn_wmma_f32_16x16x32_f16(false, a, false, b, (short)0, c, false, false);
}

static __device__ __forceinline__ void lds_wait() {
  asm volatile("s_wait_dscnt 0" ::: "memory");
}

static __device__ __forceinline__ float rcp_(float x) {
  return __builtin_amdgcn_rcpf(x);
}

#if __has_builtin(__builtin_amdgcn_tanhf)
// gfx1250 has native V_TANH_F32 (transcendental unit)
static __device__ __forceinline__ float tanh_(float x) {
  return __builtin_amdgcn_tanhf(x);
}
static __device__ __forceinline__ float sigmoid_(float x) {
  return fmaf(0.5f, __builtin_amdgcn_tanhf(0.5f * x), 0.5f);  // 1 trans op
}
#else
static __device__ __forceinline__ float tanh_(float x) {
  float xc = fminf(fmaxf(x, -15.0f), 15.0f);
  float e = __expf(2.0f * xc);
  return (e - 1.0f) * rcp_(e + 1.0f);
}
static __device__ __forceinline__ float sigmoid_(float x) {
  return rcp_(1.0f + __expf(-x));
}
#endif

static __device__ __forceinline__ float elu_(float x) {
  return (x > 0.0f) ? x : (__expf(x) - 1.0f);
}

// Load a pre-swizzled B fragment from LDS: 16 contiguous halves per lane.
static __device__ __forceinline__ v16h ldsFrag(const _Float16* p) {
  union { v16h v; v8h h[2]; } u;
  u.h[0] = *(const v8h*)(p);
  u.h[1] = *(const v8h*)(p + 8);
  return u.v;
}

// A-layout fragment (16-bit A 16x32): lane l holds row M=l&15;
// halves 0-15 -> K = kblk + 2*base + i (contiguous),
// base = (l>=16) ? 8 : 0.  Row stride in halves given by `stride`.
static __device__ __forceinline__ v16h ldsAFrag(const _Float16* p, int M, int base,
                                                int kblk, int stride) {
  union { v16h v; v8h h[2]; } u;
  u.h[0] = *(const v8h*)(p + M * stride + kblk + 2 * base);
  u.h[1] = *(const v8h*)(p + M * stride + kblk + 2 * base + 8);
  return u.v;
}

// B fragment from row-major f32 W[nRows][nCols]: B[k][n] = W[n][k].
// Fast path: K-range statically in-bounds -> 4x float4 (global_load_b128).
static __device__ __forceinline__ v16h gWFragFull(const float* __restrict__ W,
                                                  int nRows, int nCols,
                                                  int n, int k0, int lane) {
  v16h r;
  int kb = k0 + ((lane >> 4) << 4);
  if (n < nRows) {
    const v4f* p = (const v4f*)(W + n * nCols + kb);
    v4f a = p[0], b = p[1], c = p[2], d = p[3];
#pragma unroll
    for (int i = 0; i < 4; ++i) {
      r[i]      = (_Float16)a[i];
      r[4 + i]  = (_Float16)b[i];
      r[8 + i]  = (_Float16)c[i];
      r[12 + i] = (_Float16)d[i];
    }
  } else {
#pragma unroll
    for (int i = 0; i < 16; ++i) r[i] = (_Float16)0.f;
  }
  return r;
}

// Tail path with per-element K guard (only W2's last K tile needs it).
static __device__ __forceinline__ v16h gWFragTail(const float* __restrict__ W,
                                                  int nRows, int nCols,
                                                  int n, int k0, int lane) {
  v16h r;
  int kb = k0 + ((lane >> 4) << 4);
#pragma unroll
  for (int i = 0; i < 16; ++i) {
    int k = kb + i;
    float v = 0.0f;
    if (n < nRows && k < nCols) v = W[n * nCols + k];
    r[i] = (_Float16)v;
  }
  return r;
}

__global__ __launch_bounds__(256) void gru_mrf_fused_kernel(
    const float* __restrict__ X_fa_in, const float* __restrict__ fa_vals,
    const float* __restrict__ TR_vals,
    const float* __restrict__ Wih, const float* __restrict__ Whh,
    const float* __restrict__ bih, const float* __restrict__ bhh,
    const float* __restrict__ wScoreT, const float* __restrict__ wScoreM,
    const float* __restrict__ T_W1, const float* __restrict__ T_b1,
    const float* __restrict__ T_W2, const float* __restrict__ T_b2,
    const float* __restrict__ T_W3, const float* __restrict__ T_b3,
    const float* __restrict__ M_W1, const float* __restrict__ M_b1,
    const float* __restrict__ M_W2, const float* __restrict__ M_b2,
    const float* __restrict__ M_W3, const float* __restrict__ M_b3,
    float* __restrict__ out) {
  __shared__ float sT10v[128];
  __shared__ float sM0v[128];
  // Phase 1: [0,36*512) = GRU weight B-frags, [36*512, +8*16*64) = h staging.
  // Phase 2: reused as per-wave activation staging [8][16][224].
  __shared__ __align__(16) _Float16 sBig[8 * 16 * 224];

  const int tid = threadIdx.x;
  const int lane = tid & 31;
  const int wave = tid >> 5;
  const int ln = lane & 15;          // N column / A row within tile
  const int hi8 = (lane >> 4) << 3;  // 0 or 8 (C-layout row offset / A K-base)
  const int rowBase = wave * 16;
  const int b0 = blockIdx.x * 128;

  _Float16* wB = sBig;                       // 36 frags * 512 halves
  _Float16* hstage = sBig + 36 * 512;        // 8 waves * [16][64]
  _Float16* hw = hstage + wave * (16 * 64);

  // ---- cooperative pre-swizzle of GRU weights into B fragments ----
  // frag f = g*3 + kb : g in 0..11 (gate tiles), kb 0..1 = Whh K-blocks, kb 2 = Wih frag
  for (int idx = tid; idx < 36 * 512; idx += 256) {
    int f = idx >> 9;
    int r = idx & 511;
    int l = r >> 4;
    int i = r & 15;
    int g = f / 3, kb = f - g * 3;
    int n = g * 16 + (l & 15);
    float v = 0.0f;
    if (kb < 2) {
      int k = kb * 32 + ((l >> 4) << 4) + i;   // lanes 0-15: K 0-15, lanes 16-31: K 16-31
      v = Whh[n * 64 + k];
    } else {
      if (l < 16 && i < 2) v = Wih[n * 2 + i]; // K=64 -> Xf weight, K=65 -> ff weight
    }
    wB[idx] = (_Float16)v;
  }
  __syncthreads();

  // Per-lane loop-invariant scalars: score weights + gate biases (registers)
  float wTl[4], wMl[4], bRr[4], bZr[4], bNhr[4], bNir[4];
#pragma unroll
  for (int gt = 0; gt < 4; ++gt) {
    int c0 = gt * 16 + ln;
    wTl[gt] = wScoreT[c0];
    wMl[gt] = wScoreM[c0];
    bRr[gt] = bih[c0] + bhh[c0];
    bZr[gt] = bih[64 + c0] + bhh[64 + c0];
    bNir[gt] = bih[128 + c0];
    bNhr[gt] = bhh[128 + c0];
  }

  // Persistent per-lane state (C layout: row = hi8 + j, col = gt*16 + ln)
  float hC[4][8];
  float accT[4][8], accM[4][8];
  float sT[8], sM[8];
#pragma unroll
  for (int gt = 0; gt < 4; ++gt)
#pragma unroll
    for (int j = 0; j < 8; ++j) { hC[gt][j] = 0.f; accT[gt][j] = 0.f; accM[gt][j] = 0.f; }
#pragma unroll
  for (int j = 0; j < 8; ++j) { sT[j] = 0.f; sM[j] = 0.f; }

  v16h A0, A1;
#pragma unroll
  for (int i = 0; i < 16; ++i) { A0[i] = (_Float16)0.f; A1[i] = (_Float16)0.f; }

  // Per-lane input streams (lanes 16-31 duplicate lanes 0-15; masked out of A2)
  const float* pX = X_fa_in + (size_t)(b0 + rowBase + ln) * T_STEPS;
  const float* pF = fa_vals + (size_t)(b0 + rowBase + ln) * T_STEPS;
  float xv = pX[0];
  float fv = pF[0];
  float lastX = 0.f, lastF = 0.f;  // forward-fill state

#pragma unroll 1
  for (int t = 0; t < T_STEPS; ++t) {
    // ---- forward fill (segmented LOCF) ----
    if (xv != 0.0f) { lastX = xv; lastF = fv; }
    float Xf = (lane < 16) ? lastX : 0.0f;  // v_cndmask, no divergence
    float Ff = (lane < 16) ? lastF : 0.0f;
    // A2 fragment: K=64 -> Xf, K=65 -> ff, rest 0 (lanes 16-31 fully zero)
    v16h A2;
#pragma unroll
    for (int i = 0; i < 16; ++i) A2[i] = (_Float16)0.f;
    A2[0] = (_Float16)Xf;
    A2[1] = (_Float16)Ff;

    // software prefetch of next step's inputs; latency hidden under the WMMAs
    if (t + 1 < T_STEPS) {
      xv = pX[t + 1];
      fv = pF[t + 1];
    }

    // ---- GRU cell: 36 WMMAs per step per wave, zero-seeded accumulators ----
#pragma unroll
    for (int gt = 0; gt < 4; ++gt) {
      v8f cR, cZ, cH, cX;
#pragma unroll
      for (int j = 0; j < 8; ++j) { cR[j] = 0.f; cZ[j] = 0.f; cH[j] = 0.f; cX[j] = 0.f; }
      const int gz = 4 + gt, gn = 8 + gt;
      cR = wmma_f16(A0, ldsFrag(wB + (gt * 3 + 0) * 512 + lane * 16), cR);
      cR = wmma_f16(A1, ldsFrag(wB + (gt * 3 + 1) * 512 + lane * 16), cR);
      cR = wmma_f16(A2, ldsFrag(wB + (gt * 3 + 2) * 512 + lane * 16), cR);
      cZ = wmma_f16(A0, ldsFrag(wB + (gz * 3 + 0) * 512 + lane * 16), cZ);
      cZ = wmma_f16(A1, ldsFrag(wB + (gz * 3 + 1) * 512 + lane * 16), cZ);
      cZ = wmma_f16(A2, ldsFrag(wB + (gz * 3 + 2) * 512 + lane * 16), cZ);
      cH = wmma_f16(A0, ldsFrag(wB + (gn * 3 + 0) * 512 + lane * 16), cH);
      cH = wmma_f16(A1, ldsFrag(wB + (gn * 3 + 1) * 512 + lane * 16), cH);
      cX = wmma_f16(A2, ldsFrag(wB + (gn * 3 + 2) * 512 + lane * 16), cX);
#pragma unroll
      for (int j = 0; j < 8; ++j) {
        float r = sigmoid_(cR[j] + bRr[gt]);
        float z = sigmoid_(cZ[j] + bZr[gt]);
        float n = tanh_(cX[j] + bNir[gt] + r * (cH[j] + bNhr[gt]));
        hC[gt][j] = n + z * (hC[gt][j] - n);   // (1-z)n + zh
      }
    }

    // ---- attention pooling: plain exp accumulation (scores are bounded) ----
#pragma unroll
    for (int j = 0; j < 8; ++j) {
      float pT = 0.f, pM = 0.f;
#pragma unroll
      for (int gt = 0; gt < 4; ++gt) { pT += hC[gt][j] * wTl[gt]; pM += hC[gt][j] * wMl[gt]; }
      pT += __shfl_xor(pT, 1, 32); pT += __shfl_xor(pT, 2, 32);
      pT += __shfl_xor(pT, 4, 32); pT += __shfl_xor(pT, 8, 32);
      pM += __shfl_xor(pM, 1, 32); pM += __shfl_xor(pM, 2, 32);
      pM += __shfl_xor(pM, 4, 32); pM += __shfl_xor(pM, 8, 32);
      float eT = __expf(pT);
      float eM = __expf(pM);
      sT[j] += eT;
      sM[j] += eM;
#pragma unroll
      for (int gt = 0; gt < 4; ++gt) {
        accT[gt][j] += eT * hC[gt][j];
        accM[gt][j] += eM * hC[gt][j];
      }
    }

    // ---- C-layout -> A-layout transpose through per-wave LDS staging ----
    if (t + 1 < T_STEPS) {
#pragma unroll
      for (int gt = 0; gt < 4; ++gt)
#pragma unroll
        for (int j = 0; j < 8; ++j)
          hw[(hi8 + j) * 64 + gt * 16 + ln] = (_Float16)hC[gt][j];
      lds_wait();
      A0 = ldsAFrag(hw, ln, hi8, 0, 64);
      A1 = ldsAFrag(hw, ln, hi8, 32, 64);
      lds_wait();
    }
  }

  // ---- normalize attention pools IN PLACE -> accT/accM become hT/hM ----
#pragma unroll
  for (int j = 0; j < 8; ++j) {
    float iT = rcp_(sT[j]), iM = rcp_(sM[j]);
#pragma unroll
    for (int gt = 0; gt < 4; ++gt) { accT[gt][j] *= iT; accM[gt][j] *= iM; }
  }

  __syncthreads();  // repurpose sBig as per-wave activation staging [16][224]
  _Float16* aw = sBig + wave * (16 * 224);

  // ---- MLP heads via WMMA (64->200 ELU ->200 ELU ->1), both branches ----
#pragma unroll 1
  for (int br = 0; br < 2; ++br) {
    const float* W1 = br ? M_W1 : T_W1;  const float* b1 = br ? M_b1 : T_b1;
    const float* W2 = br ? M_W2 : T_W2;  const float* b2 = br ? M_b2 : T_b2;
    const float* W3 = br ? M_W3 : T_W3;  const float* b3 = br ? M_b3 : T_b3;

    // stage pooled h (C layout) into activation buffer, cols 0..63
#pragma unroll
    for (int gt = 0; gt < 4; ++gt)
#pragma unroll
      for (int j = 0; j < 8; ++j) {
        float v = br ? accM[gt][j] : accT[gt][j];
        aw[(hi8 + j) * 224 + gt * 16 + ln] = (_Float16)v;
      }
    lds_wait();
    v16h Ain0 = ldsAFrag(aw, ln, hi8, 0, 224);
    v16h Ain1 = ldsAFrag(aw, ln, hi8, 32, 224);
    lds_wait();

    // layer 1: N = 208 (13 tiles), K = 64
#pragma unroll 1
    for (int nt = 0; nt < 13; ++nt) {
      int n = nt * 16 + ln;
      float bb = (n < 200) ? b1[n] : 0.0f;
      v8f c;
#pragma unroll
      for (int j = 0; j < 8; ++j) c[j] = 0.f;
      c = wmma_f16(Ain0, gWFragFull(W1, 200, 64, n, 0, lane), c);
      c = wmma_f16(Ain1, gWFragFull(W1, 200, 64, n, 32, lane), c);
#pragma unroll
      for (int j = 0; j < 8; ++j)
        aw[(hi8 + j) * 224 + n] = (_Float16)elu_(c[j] + bb);
    }
    // zero-pad cols 208..223
    for (int idx = lane; idx < 256; idx += 32) {
      int rr = idx >> 4, cc = 208 + (idx & 15);
      aw[rr * 224 + cc] = (_Float16)0.0f;
    }
    lds_wait();
    v16h Aact[7];
#pragma unroll
    for (int kt = 0; kt < 7; ++kt) Aact[kt] = ldsAFrag(aw, ln, hi8, kt * 32, 224);
    lds_wait();

    // layer 2 (K=224 padded) fused with layer-3 dot product
    float part[8];
#pragma unroll
    for (int j = 0; j < 8; ++j) part[j] = 0.0f;
#pragma unroll 1
    for (int nt = 0; nt < 13; ++nt) {
      int n = nt * 16 + ln;
      float bb = (n < 200) ? b2[n] : 0.0f;
      v8f c;
#pragma unroll
      for (int j = 0; j < 8; ++j) c[j] = 0.f;
#pragma unroll
      for (int kt = 0; kt < 7; ++kt) {
        if (kt * 32 + 32 <= 200)   // constant-folded per unrolled kt
          c = wmma_f16(Aact[kt], gWFragFull(W2, 200, 200, n, kt * 32, lane), c);
        else
          c = wmma_f16(Aact[kt], gWFragTail(W2, 200, 200, n, kt * 32, lane), c);
      }
      float w3 = (n < 200) ? W3[n] : 0.0f;
#pragma unroll
      for (int j = 0; j < 8; ++j) part[j] += elu_(c[j] + bb) * w3;
    }
    float b3v = b3[0];
#pragma unroll
    for (int j = 0; j < 8; ++j) {
      float v = part[j];
      v += __shfl_xor(v, 1, 32); v += __shfl_xor(v, 2, 32);
      v += __shfl_xor(v, 4, 32); v += __shfl_xor(v, 8, 32);
      float raw = v + b3v;
      float sg = sigmoid_(raw);
      float val = br ? (sg * 10000.0f) : (0.1f + sg * 4.9f);
      if (ln == 0) {
        int row = hi8 + j;
        int gb = b0 + rowBase + row;
        out[(size_t)B_TOTAL * T_STEPS + (size_t)br * B_TOTAL + gb] = val;
        if (br) sM0v[rowBase + row] = val; else sT10v[rowBase + row] = val;
      }
    }
  }
  lds_wait();

  // ---- SPGR signal-equation epilogue: X_out[b][t] ----
#pragma unroll 1
  for (int i = 0; i < 17; ++i) {
    int e = lane + 32 * i;            // 0..543 = 16 rows * 34 t
    int row = e / 34;
    int t = e - row * 34;
    int gb = b0 + rowBase + row;
    float fa = fa_vals[gb * T_STEPS + t];
    float TR = TR_vals[gb];
    float T10 = sT10v[rowBase + row];
    float M0v = sM0v[rowBase + row];
    float E = __expf(-TR * rcp_(T10));
    float X = (1.0f - E) * __sinf(fa) * rcp_(1.0f - __cosf(fa) * E) * M0v;
    out[(size_t)gb * T_STEPS + t] = X;
  }
}

extern "C" void kernel_launch(void* const* d_in, const int* in_sizes, int n_in,
                              void* d_out, int out_size, void* d_ws, size_t ws_size,
                              hipStream_t stream) {
  (void)in_sizes; (void)n_in; (void)out_size; (void)d_ws; (void)ws_size;
  const float* X_fa_in = (const float*)d_in[0];
  const float* fa_vals = (const float*)d_in[1];
  // d_in[2] = fa_mask (unused), d_in[3] = fa_len (unused)
  const float* TR_vals = (const float*)d_in[4];
  const float* Wih = (const float*)d_in[5];
  const float* Whh = (const float*)d_in[6];
  const float* bih = (const float*)d_in[7];
  const float* bhh = (const float*)d_in[8];
  const float* wScoreT = (const float*)d_in[9];
  const float* wScoreM = (const float*)d_in[10];
  const float* T_W1 = (const float*)d_in[11];
  const float* T_b1 = (const float*)d_in[12];
  const float* T_W2 = (const float*)d_in[13];
  const float* T_b2 = (const float*)d_in[14];
  const float* T_W3 = (const float*)d_in[15];
  const float* T_b3 = (const float*)d_in[16];
  const float* M_W1 = (const float*)d_in[17];
  const float* M_b1 = (const float*)d_in[18];
  const float* M_W2 = (const float*)d_in[19];
  const float* M_b2 = (const float*)d_in[20];
  const float* M_W3 = (const float*)d_in[21];
  const float* M_b3 = (const float*)d_in[22];
  float* out = (float*)d_out;

  dim3 grid(B_TOTAL / 128);   // 512 blocks, 128 rows each
  dim3 block(256);            // 8 waves (wave32) x 16-row WMMA tiles
  gru_mrf_fused_kernel<<<grid, block, 0, stream>>>(
      X_fa_in, fa_vals, TR_vals, Wih, Whh, bih, bhh, wScoreT, wScoreM,
      T_W1, T_b1, T_W2, T_b2, T_W3, T_b3,
      M_W1, M_b1, M_W2, M_b2, M_W3, M_b3, out);
}
